// KGramMLPSeqModel_3015067041871
// MI455X (gfx1250) — matrix operand order repaired and checked by reading, not compiled, not run on hardware
//
#include <hip/hip_runtime.h>
#include <hip/hip_bf16.h>
#include <stddef.h>

// Problem constants (from the reference)
#define KCTX   3
#define VOCAB  32000
#define EMBED  1024
#define SEQL   256
#define BATCH  8
#define MROWS  (SEQL * BATCH)      // 2048

// GEMM tiling
#define MTILE  128                 // per workgroup (8 waves x 16 rows)
#define NTILE  64                  // per workgroup (4 x 16-wide wmma subtiles)
#define KSTEP  32                  // wmma bf16 K
#define NITER  (EMBED / KSTEP)     // 32
#define BPAD   40                  // LDS row pitch in bf16 elements (80 B, 16B-aligned)

typedef __attribute__((ext_vector_type(16))) __bf16 v16bf;
typedef __attribute__((ext_vector_type(8)))  __bf16 v8bf;
typedef __attribute__((ext_vector_type(4)))  __bf16 v4bf;
typedef __attribute__((ext_vector_type(8)))  float  v8f;
typedef __attribute__((ext_vector_type(4)))  float  v4f;
typedef __attribute__((ext_vector_type(4)))  int    v4i;

union ABFrag { v16bf v; v8bf h[2]; };

// ---- CDNA5 async global->LDS path (ASYNCcnt), with safe fallbacks ---------
#define AS1 __attribute__((address_space(1)))
#define AS3 __attribute__((address_space(3)))

#if defined(__has_builtin)
# if __has_builtin(__builtin_amdgcn_global_load_async_to_lds_b128)
#  define USE_ASYNC_LDS 1
# endif
# if __has_builtin(__builtin_amdgcn_s_wait_asynccnt)
#  define HAVE_WAIT_ASYNC_BUILTIN 1
# endif
#endif
#ifndef USE_ASYNC_LDS
# define USE_ASYNC_LDS 0
#endif

__device__ __forceinline__ void async_copy_b128(void* lds_dst, const void* gsrc) {
#if USE_ASYNC_LDS
    // prototype (from hipcc diagnostic): params are pointers to int __vector(4)
    __builtin_amdgcn_global_load_async_to_lds_b128(
        (AS1 v4i*)gsrc, (AS3 v4i*)lds_dst, /*offset=*/0, /*cpol=*/0);
#else
    (void)lds_dst; (void)gsrc;
#endif
}

__device__ __forceinline__ void wait_async0() {
#ifdef HAVE_WAIT_ASYNC_BUILTIN
    __builtin_amdgcn_s_wait_asynccnt(0);
#else
    asm volatile("s_wait_asynccnt 0x0" ::: "memory");
#endif
}

// ---------------------------------------------------------------------------
// Kernel 1: h[m,:] = relu(b1 + sum_j W1[j*VOCAB + tok_j(m)]), stored bf16.
// One block per output row m = s*BATCH + b; 256 threads x 4 floats = 1024.
// W1 rows are a gather from a 384 MB table with no reuse -> nontemporal.
// ---------------------------------------------------------------------------
__global__ __launch_bounds__(256) void build_h(const int* __restrict__ toks,
                                               const float* __restrict__ W1,
                                               const float* __restrict__ b1,
                                               __bf16* __restrict__ hbuf) {
    const int m = blockIdx.x;            // 0..2047
    const int s = m >> 3;
    const int b = m & 7;
    const int e = threadIdx.x * 4;

    // previous-K token ids (padded region -> token 0)
    const int i0 = s - 3, i1 = s - 2, i2 = s - 1;
    const int tok0 = (i0 >= 0) ? toks[i0 * BATCH + b] : 0;
    const int tok1 = (i1 >= 0) ? toks[i1 * BATCH + b] : 0;
    const int tok2 = (i2 >= 0) ? toks[i2 * BATCH + b] : 0;

    const v4f r0 = __builtin_nontemporal_load(
        (const v4f*)(W1 + ((size_t)(0 * VOCAB + tok0)) * EMBED + e));
    const v4f r1 = __builtin_nontemporal_load(
        (const v4f*)(W1 + ((size_t)(1 * VOCAB + tok1)) * EMBED + e));
    const v4f r2 = __builtin_nontemporal_load(
        (const v4f*)(W1 + ((size_t)(2 * VOCAB + tok2)) * EMBED + e));
    const v4f bb = *(const v4f*)(b1 + e);

    v4f h = bb + r0 + r1 + r2;
    v4bf o;
#pragma unroll
    for (int i = 0; i < 4; ++i) {
        const float x = h[i] > 0.f ? h[i] : 0.f;
        o[i] = (__bf16)x;
    }
    *(v4bf*)(hbuf + (size_t)m * EMBED + e) = o;
}

// ---------------------------------------------------------------------------
// Kernel 2: W2 (EMBED x VOCAB, f32, row-major) -> W2t (VOCAB x EMBED, bf16)
// 32x32 tiles via padded LDS. W2 is read exactly once (128 MB) -> nontemporal
// loads; W2t (65 MB) is re-read 16x by the GEMM -> keep it cacheable in L2.
// ---------------------------------------------------------------------------
__global__ __launch_bounds__(256) void convert_w2(const float* __restrict__ W2,
                                                  __bf16* __restrict__ W2t) {
    __shared__ float tile[32][33];
    const int v0 = blockIdx.x * 32;      // vocab base
    const int e0 = blockIdx.y * 32;      // embed base
    const int tx = threadIdx.x & 31;
    const int ty = threadIdx.x >> 5;     // 0..7

#pragma unroll
    for (int i = 0; i < 4; ++i) {
        const int e = e0 + ty + i * 8;
        tile[ty + i * 8][tx] =
            __builtin_nontemporal_load(W2 + (size_t)e * VOCAB + v0 + tx);
    }
    __syncthreads();
#pragma unroll
    for (int i = 0; i < 4; ++i) {
        const int v = v0 + ty + i * 8;
        W2t[(size_t)v * EMBED + e0 + tx] = (__bf16)tile[tx][ty + i * 8];
    }
}

// ---------------------------------------------------------------------------
// Kernel 3: out[M,N] = h[M,K] @ W2t[N,K]^T + b2, bf16 WMMA, f32 accumulate.
// Block: 256 threads = 8 waves; tile 128(M) x 64(N).
// B is staged with GLOBAL_LOAD_ASYNC_TO_LDS_B128 into a double-buffered LDS
// tile: one barrier per K-step, next fill overlaps current WMMAs.
// A fragment (16-bit A 16x32, ISA 7.12.2): lane m<16 holds row m K=0..7 &
// 16..23; lane m+16 holds K=8..15 & 24..31.
// B fragment: lane l holds column l%16; K=0..7 & 16..23 (l<16) or 8..15 & 24..31, as A.
// ---------------------------------------------------------------------------
__global__ __launch_bounds__(256) void gemm_wmma(const __bf16* __restrict__ A,
                                                 const __bf16* __restrict__ Bt,
                                                 const float* __restrict__ b2,
                                                 float* __restrict__ out) {
    __shared__ __bf16 ldsB[2][NTILE * BPAD];   // 2 x 5 KB

    const int tid  = threadIdx.x;
    const int wave = tid >> 5;
    const int lane = tid & 31;
    const int l16  = lane & 15;
    const int hi   = lane >> 4;            // 0 or 1 (K-half select)

    const int n0 = blockIdx.x * NTILE;
    const int m0 = blockIdx.y * MTILE + wave * 16;

    // cooperative B-fill indexing: 64 rows x 64 B, 256 threads x 16 B each
    const int fRow = tid >> 2;             // 0..63 (vocab row within tile)
    const int fCh  = tid & 3;              // 16-byte chunk within the 64 B row
    const __bf16* bSrc = Bt + (size_t)(n0 + fRow) * EMBED + fCh * 8;
    __bf16* bDst0 = &ldsB[0][fRow * BPAD + fCh * 8];
    __bf16* bDst1 = &ldsB[1][fRow * BPAD + fCh * 8];

    const __bf16* aRow = A + (size_t)(m0 + l16) * EMBED + hi * 8;

    v8f acc[4] = {v8f{}, v8f{}, v8f{}, v8f{}};

#if USE_ASYNC_LDS
    // prologue: kick off first B tile
    async_copy_b128(bDst0, bSrc);

    for (int it = 0; it < NITER; ++it) {
        const int k0 = it * KSTEP;
        wait_async0();                 // this wave's fill of buf[it&1] landed
        __syncthreads();               // everyone's fill landed; prev reads done
        if (it + 1 < NITER)            // overlap next fill with current WMMAs
            async_copy_b128((it & 1) ? bDst0 : bDst1, bSrc + (it + 1) * KSTEP);

        ABFrag a;
        a.h[0] = *(const v8bf*)(aRow + k0);
        a.h[1] = *(const v8bf*)(aRow + k0 + 16);
        if (it + 1 < NITER)
            __builtin_prefetch(aRow + k0 + KSTEP, 0, 1);  // global_prefetch_b8

        const __bf16* rb = (it & 1) ? &ldsB[1][0] : &ldsB[0][0];
        const __bf16* bp = rb + l16 * BPAD + hi * 8;

        ABFrag bf[4];
#pragma unroll
        for (int sub = 0; sub < 4; ++sub) {      // gather all DS loads first
            bf[sub].h[0] = *(const v8bf*)(bp + sub * 16 * BPAD);
            bf[sub].h[1] = *(const v8bf*)(bp + sub * 16 * BPAD + 16);
        }
#pragma unroll
        for (int sub = 0; sub < 4; ++sub)        // then back-to-back WMMAs
            acc[sub] = __builtin_amdgcn_wmma_f32_16x16x32_bf16(
                false, a.v, false, bf[sub].v, (short)0, acc[sub], false, false);
    }
#else
    // fallback: synchronous VGPR round-trip staging (single buffer)
    for (int k0 = 0; k0 < EMBED; k0 += KSTEP) {
        const v8bf bt = *(const v8bf*)(bSrc + k0);
        __syncthreads();
        *(v8bf*)bDst0 = bt;
        __syncthreads();

        ABFrag a;
        a.h[0] = *(const v8bf*)(aRow + k0);
        a.h[1] = *(const v8bf*)(aRow + k0 + 16);
        if (k0 + KSTEP < EMBED)
            __builtin_prefetch(aRow + k0 + KSTEP, 0, 1);

        const __bf16* bp = &ldsB[0][l16 * BPAD + hi * 8];
        ABFrag bf[4];
#pragma unroll
        for (int sub = 0; sub < 4; ++sub) {
            bf[sub].h[0] = *(const v8bf*)(bp + sub * 16 * BPAD);
            bf[sub].h[1] = *(const v8bf*)(bp + sub * 16 * BPAD + 16);
        }
#pragma unroll
        for (int sub = 0; sub < 4; ++sub)
            acc[sub] = __builtin_amdgcn_wmma_f32_16x16x32_bf16(
                false, a.v, false, bf[sub].v, (short)0, acc[sub], false, false);
    }
#endif

    // epilogue: C/D layout (32-bit 16x16): VGPR v -> M = v + 8*hi, N = l16.
    // Output (256 MB) is write-once stream -> nontemporal stores keep L2 for
    // W2t/h which are the actually re-read working set.
    float bias[4];
#pragma unroll
    for (int sub = 0; sub < 4; ++sub) bias[sub] = b2[n0 + sub * 16 + l16];

#pragma unroll
    for (int sub = 0; sub < 4; ++sub) {
        const int n = n0 + sub * 16 + l16;
#pragma unroll
        for (int v = 0; v < 8; ++v) {
            const int m = m0 + v + hi * 8;
            __builtin_nontemporal_store(acc[sub][v] + bias[sub],
                                        out + (size_t)m * VOCAB + n);
        }
    }
}

// ---------------------------------------------------------------------------
// Host launcher
// ---------------------------------------------------------------------------
extern "C" void kernel_launch(void* const* d_in, const int* in_sizes, int n_in,
                              void* d_out, int out_size, void* d_ws, size_t ws_size,
                              hipStream_t stream) {
    (void)in_sizes; (void)n_in; (void)out_size; (void)ws_size;

    const int*   toks = (const int*)d_in[0];
    const float* W1   = (const float*)d_in[1];
    const float* b1   = (const float*)d_in[2];
    const float* W2   = (const float*)d_in[3];
    const float* b2   = (const float*)d_in[4];
    float*       out  = (float*)d_out;

    // workspace layout: [ h bf16 (2048*1024) | W2t bf16 (32000*1024) ]
    __bf16* hbuf = (__bf16*)d_ws;
    __bf16* w2t  = (__bf16*)((char*)d_ws + (size_t)MROWS * EMBED * sizeof(__bf16));

    build_h<<<MROWS, 256, 0, stream>>>(toks, W1, b1, hbuf);
    convert_w2<<<dim3(VOCAB / 32, EMBED / 32), 256, 0, stream>>>(W2, w2t);
    gemm_wmma<<<dim3(VOCAB / NTILE, MROWS / MTILE), 256, 0, stream>>>(hbuf, w2t, b2, out);
}
